// TritonSelfAttention_52699248722008
// MI455X (gfx1250) — compile-verified, matrix operand order checked by reading, not run on hardware
//
#include <hip/hip_runtime.h>

typedef __attribute__((ext_vector_type(16))) _Float16 v16h;
typedef __attribute__((ext_vector_type(8)))  _Float16 v8h;
typedef __attribute__((ext_vector_type(8)))  float    v8f;
typedef int i4 __attribute__((vector_size(16)));

static constexpr int S_  = 2048;
static constexpr int H_  = 2048;
static constexpr int HD  = 128;
static constexpr int NH  = 16;
static constexpr int H3  = 6144;
static constexpr int KSTEPS = H_ / 32;

#define AS1 __attribute__((address_space(1)))
#define AS3 __attribute__((address_space(3)))

// 16-byte global -> LDS async copy (ASYNCcnt-tracked on gfx1250)
__device__ __forceinline__ void async_cp16(_Float16* lds, const _Float16* g) {
#if __has_builtin(__builtin_amdgcn_global_load_async_to_lds_b128)
  __builtin_amdgcn_global_load_async_to_lds_b128((AS1 i4*)g, (AS3 i4*)lds, 0, 0);
#else
  *(v8h*)lds = *(const v8h*)g;
#endif
}

__device__ __forceinline__ void wait_async0() {
#if __has_builtin(__builtin_amdgcn_s_wait_asynccnt)
  __builtin_amdgcn_s_wait_asynccnt(0);
#else
  asm volatile("s_wait_asynccnt 0" ::: "memory");
#endif
}

// Load a 16x32 f16 WMMA A/B fragment from a row-major tile.
// lanes 0-15 -> row = lane,    K = {0..7, 16..23}
// lanes16-31 -> row = lane-16, K = {8..15, 24..31}
__device__ __forceinline__ v16h load_frag(const _Float16* base, int stride, int lane) {
  const int half = (lane >> 4) & 1;
  const _Float16* p = base + (lane & 15) * stride + half * 8;
  v8h lo = *(const v8h*)(p);
  v8h hi = *(const v8h*)(p + 16);
  return __builtin_shufflevector(lo, hi, 0,1,2,3,4,5,6,7,8,9,10,11,12,13,14,15);
}

__device__ __forceinline__ v8f wmma16(v16h a, v16h b, v8f c) {
  return __builtin_amdgcn_wmma_f32_16x16x32_f16(false, a, false, b, (short)0, c,
                                                false, false);
}

// ---------------------------------------------------------------------------
// Kernel 1: QKV = x @ w_qkv + b. 128x128 tile / 256 threads / 8 waves.
// Double-buffered LDS; next tile staged in registers during compute.
// Writes Q,K as [bh][S][HD] f16, V transposed as [bh][HD][S] f16.
// ---------------------------------------------------------------------------
__global__ void __launch_bounds__(256)
qkv_gemm(const float* __restrict__ x, const float* __restrict__ w,
         const float* __restrict__ bias,
         _Float16* __restrict__ qws, _Float16* __restrict__ kws,
         _Float16* __restrict__ vws) {
  __shared__ _Float16 As[2][128][40];
  __shared__ _Float16 Bs[2][128][40];
  const int tid = threadIdx.x;
  const int lane = tid & 31, wave = tid >> 5;
  const int mw = wave & 3, nw = wave >> 2;        // 4 M-groups x 2 N-groups
  const int n0 = blockIdx.x * 128, m0 = blockIdx.y * 128;
  v8f acc[2][4] = {};

  const int arow = tid >> 1, acs = (tid & 1) * 16;
  const int bn = tid & 127, bkh = (tid >> 7) * 16;
  const float* asrc = x + (size_t)(m0 + arow) * H_ + acs;

  float areg[16], breg[16];
  auto ldA = [&](int k0) {
#pragma unroll
    for (int j = 0; j < 16; ++j) areg[j] = asrc[k0 + j];
  };
  auto ldB = [&](int k0) {
    const float* s = w + (size_t)(k0 + bkh) * H3 + n0 + bn;
    if (k0 + 32 < H_) __builtin_prefetch(s + (size_t)32 * H3, 0, 1);
#pragma unroll
    for (int j = 0; j < 16; ++j) breg[j] = s[(size_t)j * H3];
  };
  auto stTiles = [&](int buf) {
    v8h h0, h1;
#pragma unroll
    for (int j = 0; j < 8; ++j) { h0[j] = (_Float16)areg[j]; h1[j] = (_Float16)areg[j + 8]; }
    *(v8h*)&As[buf][arow][acs]     = h0;
    *(v8h*)&As[buf][arow][acs + 8] = h1;
#pragma unroll
    for (int j = 0; j < 8; ++j) { h0[j] = (_Float16)breg[j]; h1[j] = (_Float16)breg[j + 8]; }
    *(v8h*)&Bs[buf][bn][bkh]     = h0;
    *(v8h*)&Bs[buf][bn][bkh + 8] = h1;
  };

  ldA(0); ldB(0);
  for (int it = 0; it < KSTEPS; ++it) {
    const int cur = it & 1;
    stTiles(cur);
    __syncthreads();
    if (it + 1 < KSTEPS) { ldA((it + 1) * 32); ldB((it + 1) * 32); }
    v16h a0 = load_frag(&As[cur][mw * 32][0], 40, lane);
    v16h a1 = load_frag(&As[cur][mw * 32 + 16][0], 40, lane);
#pragma unroll
    for (int t = 0; t < 4; ++t) {
      v16h b = load_frag(&Bs[cur][nw * 64 + t * 16][0], 40, lane);
      acc[0][t] = wmma16(a0, b, acc[0][t]);
      acc[1][t] = wmma16(a1, b, acc[1][t]);
    }
  }

  const int half = (lane >> 4) & 1;
#pragma unroll
  for (int t = 0; t < 4; ++t) {
    int col = n0 + nw * 64 + t * 16 + (lane & 15);
    int which = col >> 11, within = col & 2047;
    int hh = within >> 7, d = within & 127;
    float bv = bias[col];
#pragma unroll
    for (int mt = 0; mt < 2; ++mt)
#pragma unroll
      for (int i = 0; i < 8; ++i) {
        int row = m0 + mw * 32 + mt * 16 + i + half * 8;
        int bb = row >> 11, s = row & 2047;
        size_t bh = (size_t)bb * NH + hh;
        _Float16 hv = (_Float16)(acc[mt][t][i] + bv);
        if (which == 0)      qws[(bh * S_ + s) * HD + d] = hv;
        else if (which == 1) kws[(bh * S_ + s) * HD + d] = hv;
        else                 vws[(bh * HD + d) * S_ + s] = hv;   // V transposed
      }
  }
}

// ---------------------------------------------------------------------------
// Kernel 2: causal flash attention. One WG = 64 query rows of one (b,h);
// each wave owns 16 query rows. K/V tiles staged with async global->LDS.
// ---------------------------------------------------------------------------
__global__ void __launch_bounds__(128)
attn(const _Float16* __restrict__ qws, const _Float16* __restrict__ kws,
     const _Float16* __restrict__ vws, _Float16* __restrict__ cws) {
  __shared__ _Float16 Ks[64][136];    // K tile  [key][hd]
  __shared__ _Float16 Vts[128][72];   // V^T tile [hd][key]
  __shared__ _Float16 Ps[4][16][72];  // per-wave P transpose bounce
  const int tid = threadIdx.x, lane = tid & 31, wave = tid >> 5;
  const int half = (lane >> 4) & 1;
  const int bh = blockIdx.x, qblk = blockIdx.y;
  const int q0 = qblk * 64 + wave * 16;
  const float sc = 0.08838834764831845f;  // 1/sqrt(128)

  v16h qf[4];
#pragma unroll
  for (int c = 0; c < 4; ++c)
    qf[c] = load_frag(qws + ((size_t)bh * S_ + q0) * HD + c * 32, HD, lane);

  float m_i[8], l_i[8];
#pragma unroll
  for (int i = 0; i < 8; ++i) { m_i[i] = -1e30f; l_i[i] = 0.f; }
  v8f o[8] = {};

  const int krow = tid >> 1, kcs = (tid & 1) * 64;

  for (int j0 = 0; j0 <= qblk * 64; j0 += 64) {
    { // K tile: async global->LDS (f16, no conversion needed)
      const _Float16* src = kws + ((size_t)bh * S_ + j0 + krow) * HD + kcs;
#pragma unroll
      for (int i = 0; i < 8; ++i)
        async_cp16(&Ks[krow][kcs + i * 8], src + i * 8);
    }
    { // V^T tile: async global->LDS
      const _Float16* src = vws + ((size_t)bh * HD + tid) * S_ + j0;
#pragma unroll
      for (int i = 0; i < 8; ++i)
        async_cp16(&Vts[tid][i * 8], src + i * 8);
    }
    wait_async0();
    __syncthreads();

    // S = Q * K^T  (4 key tiles x 4 hd chunks = 16 WMMAs)
    v8f sacc[4] = {};
#pragma unroll
    for (int t = 0; t < 4; ++t) {
      v8f s = {};
#pragma unroll
      for (int c = 0; c < 4; ++c) {
        v16h b = load_frag(&Ks[t * 16][c * 32], 136, lane);
        s = wmma16(qf[c], b, s);
      }
      sacc[t] = s;
    }

    if (j0 + 64 > q0) {  // diagonal block: causal mask
#pragma unroll
      for (int t = 0; t < 4; ++t) {
        int kc = j0 + t * 16 + (lane & 15);
#pragma unroll
        for (int i = 0; i < 8; ++i)
          if (kc > q0 + i + half * 8) sacc[t][i] = -1e30f;
      }
    }

    // online softmax; rows live in 16-lane halves -> xor-shuffle reduce
    float alpha[8];
#pragma unroll
    for (int i = 0; i < 8; ++i) {
      float mx = sacc[0][i];
#pragma unroll
      for (int t = 1; t < 4; ++t) mx = fmaxf(mx, sacc[t][i]);
      for (int sft = 8; sft >= 1; sft >>= 1) mx = fmaxf(mx, __shfl_xor(mx, sft, 32));
      mx *= sc;
      float mnew = fmaxf(m_i[i], mx);
      alpha[i] = __expf(m_i[i] - mnew);
      m_i[i] = mnew;
      float rs = 0.f;
#pragma unroll
      for (int t = 0; t < 4; ++t) {
        float p = __expf(sacc[t][i] * sc - mnew);
        Ps[wave][i + half * 8][t * 16 + (lane & 15)] = (_Float16)p;
        rs += p;
      }
      for (int sft = 8; sft >= 1; sft >>= 1) rs += __shfl_xor(rs, sft, 32);
      l_i[i] = l_i[i] * alpha[i] + rs;
    }
#pragma unroll
    for (int t8 = 0; t8 < 8; ++t8)
#pragma unroll
      for (int i = 0; i < 8; ++i) o[t8][i] *= alpha[i];

    // O += P * V  (8 hd tiles x 2 key chunks = 16 WMMAs)
    v16h pf0 = load_frag(&Ps[wave][0][0], 72, lane);
    v16h pf1 = load_frag(&Ps[wave][0][32], 72, lane);
#pragma unroll
    for (int t8 = 0; t8 < 8; ++t8) {
      v16h b0 = load_frag(&Vts[t8 * 16][0], 72, lane);
      v16h b1 = load_frag(&Vts[t8 * 16][32], 72, lane);
      o[t8] = wmma16(pf0, b0, o[t8]);
      o[t8] = wmma16(pf1, b1, o[t8]);
    }
    __syncthreads();
  }

  // normalize + store ctx row-major [B][S][H] f16
  const int b = bh >> 4, h = bh & 15;
#pragma unroll
  for (int i = 0; i < 8; ++i) {
    float inv = 1.f / l_i[i];
    int s = q0 + i + half * 8;
#pragma unroll
    for (int t8 = 0; t8 < 8; ++t8) {
      int col = h * HD + t8 * 16 + (lane & 15);
      cws[((size_t)b * S_ + s) * H_ + col] = (_Float16)(o[t8][i] * inv);
    }
  }
}

// ---------------------------------------------------------------------------
// Kernel 3: out = ctx @ w_o. 128x128 tile / 8 waves; double-buffered LDS,
// A tile via async global->LDS issued one iteration ahead.
// ---------------------------------------------------------------------------
__global__ void __launch_bounds__(256)
oproj(const _Float16* __restrict__ a, const float* __restrict__ w,
      float* __restrict__ out) {
  __shared__ _Float16 As[2][128][40];
  __shared__ _Float16 Bs[2][128][40];
  const int tid = threadIdx.x, lane = tid & 31, wave = tid >> 5;
  const int mw = wave & 3, nw = wave >> 2;
  const int n0 = blockIdx.x * 128, m0 = blockIdx.y * 128;
  v8f acc[2][4] = {};

  const int arow = tid >> 1, acs = (tid & 1) * 16;
  const int bn = tid & 127, bkh = (tid >> 7) * 16;
  const _Float16* asrc = a + (size_t)(m0 + arow) * H_ + acs;

  float breg[16];
  auto ldB = [&](int k0) {
    const float* s = w + (size_t)(k0 + bkh) * H_ + n0 + bn;
    if (k0 + 32 < H_) __builtin_prefetch(s + (size_t)32 * H_, 0, 1);
#pragma unroll
    for (int j = 0; j < 16; ++j) breg[j] = s[(size_t)j * H_];
  };
  auto ldA_async = [&](int k0, int buf) {
    async_cp16(&As[buf][arow][acs],     asrc + k0);
    async_cp16(&As[buf][arow][acs + 8], asrc + k0 + 8);
  };
  auto stB = [&](int buf) {
    v8h h0, h1;
#pragma unroll
    for (int j = 0; j < 8; ++j) { h0[j] = (_Float16)breg[j]; h1[j] = (_Float16)breg[j + 8]; }
    *(v8h*)&Bs[buf][bn][bkh]     = h0;
    *(v8h*)&Bs[buf][bn][bkh + 8] = h1;
  };

  ldB(0);
  ldA_async(0, 0);
  for (int it = 0; it < KSTEPS; ++it) {
    const int cur = it & 1;
    stB(cur);
    wait_async0();          // this wave's async A copies for tile `cur` done
    __syncthreads();        // all waves' stores/copies visible; prior reads done
    if (it + 1 < KSTEPS) { ldB((it + 1) * 32); ldA_async((it + 1) * 32, cur ^ 1); }
    v16h a0 = load_frag(&As[cur][mw * 32][0], 40, lane);
    v16h a1 = load_frag(&As[cur][mw * 32 + 16][0], 40, lane);
#pragma unroll
    for (int t = 0; t < 4; ++t) {
      v16h b = load_frag(&Bs[cur][nw * 64 + t * 16][0], 40, lane);
      acc[0][t] = wmma16(a0, b, acc[0][t]);
      acc[1][t] = wmma16(a1, b, acc[1][t]);
    }
  }

  const int half = (lane >> 4) & 1;
#pragma unroll
  for (int t = 0; t < 4; ++t) {
    int col = n0 + nw * 64 + t * 16 + (lane & 15);
#pragma unroll
    for (int mt = 0; mt < 2; ++mt)
#pragma unroll
      for (int i = 0; i < 8; ++i) {
        int row = m0 + mw * 32 + mt * 16 + i + half * 8;
        out[(size_t)row * H_ + col] = acc[mt][t][i];
      }
  }
}

extern "C" void kernel_launch(void* const* d_in, const int* in_sizes, int n_in,
                              void* d_out, int out_size, void* d_ws, size_t ws_size,
                              hipStream_t stream) {
  const float* x     = (const float*)d_in[0];
  const float* w_qkv = (const float*)d_in[1];
  const float* b_qkv = (const float*)d_in[2];
  const float* w_o   = (const float*)d_in[3];
  float* out = (float*)d_out;

  const size_t NEL = (size_t)2 * NH * S_ * HD;  // 8M f16 elements per tensor
  _Float16* qws = (_Float16*)d_ws;
  _Float16* kws = qws + NEL;
  _Float16* vws = kws + NEL;
  _Float16* cws = vws + NEL;                    // total 64 MB of ws

  qkv_gemm<<<dim3(H3 / 128, (2 * S_) / 128), 256, 0, stream>>>(x, w_qkv, b_qkv,
                                                               qws, kws, vws);
  attn<<<dim3(2 * NH, S_ / 64), 128, 0, stream>>>(qws, kws, vws, cws);
  oproj<<<dim3(H_ / 128, (2 * S_) / 128), 256, 0, stream>>>(cws, w_o, out);
}
